// BoxHead_72834055406188
// MI455X (gfx1250) — compile-verified, matrix operand order checked by reading, not run on hardware
//
#include <hip/hip_runtime.h>
#include <hip/hip_bf16.h>
#include <math.h>

// ---------------------------------------------------------------------------
// Problem constants (from reference)
// ---------------------------------------------------------------------------
#define NBOX   512
#define CCH    512
#define FH     50
#define FW     50
#define POOL   7
#define SRAT   2
#define SGRID  (POOL * SRAT)          // 14
#define DIN    (CCH * POOL * POOL)    // 25088
#define FDIM   4096
#define KCLS   21
#define NCAND  (NBOX * (KCLS - 1))    // 10240
#define DETS   100
#define IMGW   800.0f
#define IMGH   800.0f
#define BBOX_CLIP 4.1351666f          // log(1000/16)

typedef __bf16 v16bf __attribute__((ext_vector_type(16)));
typedef __bf16 v4bf  __attribute__((ext_vector_type(4)));
typedef float  v8f   __attribute__((ext_vector_type(8)));

// ---------------------------------------------------------------------------
// TDM availability probe (ROCm7.2 clang-22 = 5-arg builtin; clang-23 = 6-arg)
// ---------------------------------------------------------------------------
#if defined(__has_builtin)
#if __has_builtin(__builtin_amdgcn_tensor_load_to_lds)
#define HAVE_TDM 1
#endif
#if __has_builtin(__builtin_amdgcn_s_wait_tensorcnt)
#define HAVE_TWAIT 1
#endif
#endif
#ifndef HAVE_TDM
#define HAVE_TDM 0
#endif
#ifndef HAVE_TWAIT
#define HAVE_TWAIT 0
#endif

// s_wait_tensorcnt needs a compile-time constant immediate -> template param.
template <short CNT>
__device__ __forceinline__ void tdm_wait() {
#if HAVE_TWAIT
    __builtin_amdgcn_s_wait_tensorcnt(CNT);
#endif
}

// Issue a TDM load of a 32(K-rows) x 32(cols) fp32 tile, row stride Nstride
// elements, from gsrc into LDS byte offset lds_off.  2-D descriptor per
// CDNA5 ISA (D# groups; data_size=4B, type=2).
__device__ __forceinline__ void tdm_load_tile(const float* gsrc,
                                              unsigned lds_off, int Nstride) {
#if HAVE_TDM
    typedef unsigned int v4u __attribute__((ext_vector_type(4)));
    typedef int v8i __attribute__((ext_vector_type(8)));
    typedef int v4i __attribute__((ext_vector_type(4)));
    unsigned long long ga = (unsigned long long)(size_t)gsrc;
    v4u g0;
    g0[0] = 1u;                                   // count=1 (valid descriptor)
    g0[1] = lds_off;                              // lds_addr
    g0[2] = (unsigned)ga;                         // global_addr[31:0]
    g0[3] = ((unsigned)(ga >> 32) & 0x01FFFFFFu) | (2u << 30);  // addr[56:32]|type=2
    v8i g1;
    g1[0] = (int)(2u << 16);                      // data_size=2 (4 bytes)
    g1[1] = (int)(((unsigned)Nstride & 0xFFFFu) << 16);     // tensor_dim0[15:0]
    g1[2] = (int)((((unsigned)Nstride >> 16) & 0xFFFFu) | (32u << 16)); // td0 hi | td1 lo
    g1[3] = (int)(32u << 16);                     // td1 hi=0 | tile_dim0=32
    g1[4] = 32;                                   // tile_dim1=32 | tile_dim2=0
    g1[5] = Nstride;                              // tensor_dim0_stride[31:0]
    g1[6] = 0;                                    // stride hi | td1_stride lo
    g1[7] = 0;
    v4i z4 = {0, 0, 0, 0};
#if __clang_major__ >= 23
    v8i z8 = {0, 0, 0, 0, 0, 0, 0, 0};
    __builtin_amdgcn_tensor_load_to_lds(g0, g1, z4, z4, z8, 0);
#else
    __builtin_amdgcn_tensor_load_to_lds(g0, g1, z4, z4, 0);
#endif
#endif
}

// ---------------------------------------------------------------------------
// Packed WMMA-A layout helper. A[MxK] bf16 as [M/16][K/32][lane(32)][elem(16)]
//   row   = mi*16 + (lane & 15),  khalf = lane >> 4
//   K offs: khalf*8 + {0..7} (elems 0..7), khalf*8 + 16 + {0..7} (elems 8..15)
// ---------------------------------------------------------------------------
__device__ __forceinline__ size_t apack_idx(int row, int kabs, int Kdim) {
    int mi = row >> 4;
    int lane_low = row & 15;
    int ki = kabs >> 5;
    int r  = kabs & 31;
    int khalf = (r >> 3) & 1;
    int elem  = (r & 7) | (((r >> 4) & 1) << 3);
    int lane  = khalf * 16 + lane_low;
    return ((((size_t)mi * (Kdim >> 5) + ki) << 5) + lane) * 16 + elem;
}

// ---------------------------------------------------------------------------
// Kernel 1: ROI align -> packed-layout bf16 activation matrix X
// ---------------------------------------------------------------------------
__global__ void roi_align_kernel(const float* __restrict__ feat,
                                 const float* __restrict__ props,
                                 __bf16* __restrict__ Xp) {
    __shared__ float lyv[SGRID], vyv[SGRID];
    __shared__ float lxv[SGRID], vxv[SGRID];
    __shared__ int   y0v[SGRID], y1v[SGRID], x0v[SGRID], x1v[SGRID];

    const int n = blockIdx.x;
    const float px1 = props[n * 4 + 0] * (1.0f / 16.0f);
    const float py1 = props[n * 4 + 1] * (1.0f / 16.0f);
    const float px2 = props[n * 4 + 2] * (1.0f / 16.0f);
    const float py2 = props[n * 4 + 3] * (1.0f / 16.0f);
    const float rw = fmaxf(px2 - px1, 1.0f);
    const float rh = fmaxf(py2 - py1, 1.0f);

    int t = threadIdx.x;
    if (t < SGRID) {
        float frac = (t + 0.5f) / (float)SGRID;
        float ys = py1 + rh * frac;
        float vy = (ys >= -1.0f && ys <= (float)FH) ? 1.0f : 0.0f;
        ys = fminf(fmaxf(ys, 0.0f), (float)(FH - 1));
        float y0 = floorf(ys);
        lyv[t] = ys - y0; vyv[t] = vy;
        y0v[t] = (int)y0; y1v[t] = min((int)y0 + 1, FH - 1);
        float xs = px1 + rw * frac;
        float vx = (xs >= -1.0f && xs <= (float)FW) ? 1.0f : 0.0f;
        xs = fminf(fmaxf(xs, 0.0f), (float)(FW - 1));
        float x0 = floorf(xs);
        lxv[t] = xs - x0; vxv[t] = vx;
        x0v[t] = (int)x0; x1v[t] = min((int)x0 + 1, FW - 1);
    }
    __syncthreads();

    for (int idx = threadIdx.x; idx < CCH * POOL * POOL; idx += blockDim.x) {
        int c   = idx / (POOL * POOL);
        int bin = idx - c * (POOL * POOL);
        int py  = bin / POOL;
        int pxx = bin - py * POOL;
        const float* fc = feat + (size_t)c * (FH * FW);
        float sum = 0.0f;
#pragma unroll
        for (int sy = 0; sy < SRAT; ++sy) {
            int iy = py * SRAT + sy;
            float ly = lyv[iy], hy = 1.0f - ly, vy = vyv[iy];
            int yy0 = y0v[iy], yy1 = y1v[iy];
#pragma unroll
            for (int sx = 0; sx < SRAT; ++sx) {
                int ix = pxx * SRAT + sx;
                float lx = lxv[ix], hx = 1.0f - lx, vx = vxv[ix];
                int xx0 = x0v[ix], xx1 = x1v[ix];
                float v = fc[yy0 * FW + xx0] * hy * hx
                        + fc[yy0 * FW + xx1] * hy * lx
                        + fc[yy1 * FW + xx0] * ly * hx
                        + fc[yy1 * FW + xx1] * ly * lx;
                sum += v * (vy * vx);
            }
        }
        Xp[apack_idx(n, idx, DIN)] = (__bf16)(sum * 0.25f);
    }
}

// ---------------------------------------------------------------------------
// Kernel 2: GEMM  Out = relu(A_packed_bf16[512 x K] * W_f32[K x N] + b)
// Block = 256 threads (8 waves) covers ALL 512 rows x one 32-col strip, so
// the HBM-resident weight matrix is streamed exactly once (activations are
// L2-resident). Wave w owns row-tiles {w+8j, j=0..3} x 2 col groups ->
// 8 accumulators, 8 WMMAs per 32-deep K-step.
// Weight tiles (32K x 32col fp32, 4KB) are TDM double-buffered into LDS,
// then converted to fragment-ordered bf16 for 2x ds_load_b128 consumption.
// ---------------------------------------------------------------------------
template <bool PACK_OUT>
__global__ void gemm_bf16_wmma(const __bf16* __restrict__ Apack,
                               const float* __restrict__ W,
                               const float* __restrict__ bias,
                               __bf16* __restrict__ Out,
                               int K, int N) {
    __shared__ float  wtile[2][32 * 32];      // fp32 weight tiles (TDM dest)
    __shared__ __bf16 Bfrag[2 * 32 * 16];     // fragment-ordered bf16

    const int tid   = threadIdx.x;
    const int wave  = tid >> 5;
    const int lane  = tid & 31;
    const int col16 = lane & 15;
    const int khalf = lane >> 4;

    const int n0 = blockIdx.x * 32;

    float bvals[2];
#pragma unroll
    for (int g = 0; g < 2; ++g) bvals[g] = bias[n0 + g * 16 + col16];

    v8f acc[4][2] = {};

    // Per-lane packed A stream base for each of the wave's 4 row-tiles.
    const __bf16* abase[4];
#pragma unroll
    for (int j = 0; j < 4; ++j) {
        int rt = wave + 8 * j;               // row-tile index (rows rt*16..)
        abase[j] = Apack + ((size_t)rt * (K >> 5) * 32 + lane) * 16;
    }

    // Staging role: 64 slots (2 groups x 32 lanes), 4 threads per slot.
    const int slot  = tid >> 2;              // 0..63
    const int part  = tid & 3;               // 0..3 -> elems part*4..part*4+3
    const int sg    = slot >> 5;             // col group 0..1
    const int slane = slot & 31;
    const int scol  = sg * 16 + (slane & 15);
    const int skh   = slane >> 4;
    __bf16* sdst = &Bfrag[(sg * 32 + slane) * 16 + part * 4];

    const int nk = K >> 5;

    // Prologue: prefetch first weight tile.
    if (wave == 0)
        tdm_load_tile(W + 0 * (size_t)N + n0,
                      (unsigned)(size_t)&wtile[0][0], N);

    for (int i = 0; i < nk; ++i) {
        const int cur = i & 1;
        if (wave == 0) {
            if (i + 1 < nk) {
                tdm_load_tile(W + (size_t)(i + 1) * 32 * N + n0,
                              (unsigned)(size_t)&wtile[(i + 1) & 1][0], N);
                tdm_wait<1>();   // current tile arrived (in-order completion)
            } else {
                tdm_wait<0>();   // last tile arrived
            }
        }
        __syncthreads();   // weight tile [cur] visible to all waves

        // stage: fp32 tile -> fragment-ordered bf16 (4 elems per thread)
        {
            float v[4];
#pragma unroll
            for (int q = 0; q < 4; ++q) {
                int e = part * 4 + q;
                int kofs = (e < 8) ? (skh * 8 + e) : (skh * 8 + 16 + (e - 8));
#if HAVE_TDM
                v[q] = wtile[cur][kofs * 32 + scol];
#else
                v[q] = W[(size_t)(i * 32 + kofs) * N + n0 + scol];
#endif
            }
            v4bf f4;
#pragma unroll
            for (int q = 0; q < 4; ++q) f4[q] = (__bf16)v[q];
            *(v4bf*)sdst = f4;
        }
        __syncthreads();   // fragments ready

        // B fragments (shared by all row-tiles of this wave)
        v16bf bfr[2];
#pragma unroll
        for (int g = 0; g < 2; ++g)
            bfr[g] = *(const v16bf*)&Bfrag[(g * 32 + lane) * 16];

        // A fragments + 8 WMMAs
#pragma unroll
        for (int j = 0; j < 4; ++j) {
            v16bf af = *(const v16bf*)(abase[j] + (size_t)i * 512);
#pragma unroll
            for (int g = 0; g < 2; ++g)
                acc[j][g] = __builtin_amdgcn_wmma_f32_16x16x32_bf16(
                    false, af, false, bfr[g], (short)0, acc[j][g], false, false);
        }
        __syncthreads();   // all consumers done before next tile overwrites
    }

    // Epilogue: D layout vgpr v -> M = v + 8*khalf ; N = lane&15
#pragma unroll
    for (int j = 0; j < 4; ++j) {
        int rt = wave + 8 * j;
#pragma unroll
        for (int g = 0; g < 2; ++g) {
#pragma unroll
            for (int v = 0; v < 8; ++v) {
                int row  = rt * 16 + v + 8 * khalf;
                int colK = n0 + g * 16 + col16;
                float r = fmaxf(acc[j][g][v] + bvals[g], 0.0f);
                if (PACK_OUT)
                    Out[apack_idx(row, colK, N)] = (__bf16)r;
                else
                    Out[(size_t)row * N + colK] = (__bf16)r;
            }
        }
    }
}

// ---------------------------------------------------------------------------
// Kernel 3: heads — logits[512x21], deltas[512x84]. One block per row.
// ---------------------------------------------------------------------------
__global__ void heads_kernel(const __bf16* __restrict__ H2,
                             const float* __restrict__ cls_w,
                             const float* __restrict__ cls_b,
                             const float* __restrict__ bbox_w,
                             const float* __restrict__ bbox_b,
                             float* __restrict__ logits,
                             float* __restrict__ deltas) {
    __shared__ float xrow[FDIM];
    const int row = blockIdx.x;
    for (int k = threadIdx.x; k < FDIM; k += blockDim.x)
        xrow[k] = (float)H2[(size_t)row * FDIM + k];
    __syncthreads();

    int t = threadIdx.x;
    if (t < KCLS) {
        float acc = cls_b[t];
        for (int k = 0; k < FDIM; ++k)
            acc += xrow[k] * cls_w[(size_t)k * KCLS + t];
        logits[row * KCLS + t] = acc;
    } else if (t < KCLS + KCLS * 4) {
        int c = t - KCLS;
        float acc = bbox_b[c];
        for (int k = 0; k < FDIM; ++k)
            acc += xrow[k] * bbox_w[(size_t)k * (KCLS * 4) + c];
        deltas[row * (KCLS * 4) + c] = acc;
    }
}

// ---------------------------------------------------------------------------
// Kernel 4: softmax + decode + clip + validity -> candidate arrays
// ---------------------------------------------------------------------------
__global__ void decode_kernel(const float* __restrict__ logits,
                              const float* __restrict__ deltas,
                              const float* __restrict__ props,
                              float* __restrict__ scall,
                              float* __restrict__ smask,
                              float* __restrict__ boxes,
                              float* __restrict__ obox) {
    int j = blockIdx.x * blockDim.x + threadIdx.x;
    if (j >= NCAND) return;
    int row = j / (KCLS - 1);
    int cls = (j - row * (KCLS - 1)) + 1;

    const float* lg = logits + row * KCLS;
    float m = lg[0];
    for (int i = 1; i < KCLS; ++i) m = fmaxf(m, lg[i]);
    float sum = 0.0f;
    for (int i = 0; i < KCLS; ++i) sum += expf(lg[i] - m);
    float sc = expf(lg[cls] - m) / sum;

    const float* p = props + row * 4;
    float w  = p[2] - p[0];
    float h  = p[3] - p[1];
    float cx = p[0] + 0.5f * w;
    float cy = p[1] + 0.5f * h;
    const float* d = deltas + row * (KCLS * 4) + cls * 4;
    float dx = d[0] * 0.1f;
    float dy = d[1] * 0.1f;
    float dw = fminf(d[2] * 0.2f, BBOX_CLIP);
    float dh = fminf(d[3] * 0.2f, BBOX_CLIP);
    float pcx = dx * w + cx;
    float pcy = dy * h + cy;
    float pw  = expf(dw) * w;
    float ph  = expf(dh) * h;
    float x1 = fminf(fmaxf(pcx - 0.5f * pw, 0.0f), IMGW);
    float y1 = fminf(fmaxf(pcy - 0.5f * ph, 0.0f), IMGH);
    float x2 = fminf(fmaxf(pcx + 0.5f * pw, 0.0f), IMGW);
    float y2 = fminf(fmaxf(pcy + 0.5f * ph, 0.0f), IMGH);

    float bw = x2 - x1, bh = y2 - y1;
    bool valid = (sc > 0.05f) && (bw >= 1.0f) && (bh >= 1.0f);

    scall[j] = sc;
    smask[j] = valid ? sc : -INFINITY;
    boxes[j * 4 + 0] = x1; boxes[j * 4 + 1] = y1;
    boxes[j * 4 + 2] = x2; boxes[j * 4 + 3] = y2;
    float off = (float)cls * (IMGW + 1.0f);
    obox[j * 4 + 0] = x1 + off; obox[j * 4 + 1] = y1 + off;
    obox[j * 4 + 2] = x2 + off; obox[j * 4 + 3] = y2 + off;
}

// ---------------------------------------------------------------------------
// Kernel 5: greedy NMS, single workgroup (1024 threads), scores in LDS
// out layout: boxes[100*4] | scores[100] | labels[100]  (600 floats)
// ---------------------------------------------------------------------------
__global__ void nms_kernel(const float* __restrict__ scall,
                           const float* __restrict__ smask,
                           const float* __restrict__ boxes,
                           const float* __restrict__ obox,
                           float* __restrict__ out) {
    __shared__ float ssh[NCAND];
    __shared__ float rv[1024];
    __shared__ int   ri[1024];
    __shared__ float bb[4];
    __shared__ int   bi_sh;
    __shared__ float bv_sh;

    const int t = threadIdx.x;
    for (int j = t; j < NCAND; j += 1024) ssh[j] = smask[j];
    __syncthreads();

    for (int det = 0; det < DETS; ++det) {
        float bestv = -INFINITY;
        int   besti = t;
        for (int j = t; j < NCAND; j += 1024) {
            float v = ssh[j];
            if (v > bestv) { bestv = v; besti = j; }
        }
        rv[t] = bestv; ri[t] = besti;
        __syncthreads();
        for (int off = 512; off > 0; off >>= 1) {
            if (t < off) {
                float v2 = rv[t + off]; int i2 = ri[t + off];
                if (v2 > rv[t] || (v2 == rv[t] && i2 < ri[t])) { rv[t] = v2; ri[t] = i2; }
            }
            __syncthreads();
        }
        if (t == 0) {
            bi_sh = ri[0];
            bv_sh = rv[0];
            int i = ri[0];
            bb[0] = obox[i * 4 + 0]; bb[1] = obox[i * 4 + 1];
            bb[2] = obox[i * 4 + 2]; bb[3] = obox[i * 4 + 3];
            bool found = (rv[0] > -INFINITY) && isfinite(rv[0]);
            if (found) {
                out[det * 4 + 0] = boxes[i * 4 + 0];
                out[det * 4 + 1] = boxes[i * 4 + 1];
                out[det * 4 + 2] = boxes[i * 4 + 2];
                out[det * 4 + 3] = boxes[i * 4 + 3];
                out[DETS * 4 + det] = scall[i];
                out[DETS * 5 + det] = (float)((i % (KCLS - 1)) + 1);
            } else {
                out[det * 4 + 0] = 0.0f; out[det * 4 + 1] = 0.0f;
                out[det * 4 + 2] = 0.0f; out[det * 4 + 3] = 0.0f;
                out[DETS * 4 + det] = 0.0f;
                out[DETS * 5 + det] = -1.0f;
            }
        }
        __syncthreads();

        if (isfinite(bv_sh) && bv_sh > -INFINITY) {
            float kx1 = bb[0], ky1 = bb[1], kx2 = bb[2], ky2 = bb[3];
            float a1 = (kx2 - kx1) * (ky2 - ky1);
            for (int j = t; j < NCAND; j += 1024) {
                float ox1 = obox[j * 4 + 0], oy1 = obox[j * 4 + 1];
                float ox2 = obox[j * 4 + 2], oy2 = obox[j * 4 + 3];
                float ix1 = fmaxf(kx1, ox1), iy1 = fmaxf(ky1, oy1);
                float ix2 = fminf(kx2, ox2), iy2 = fminf(ky2, oy2);
                float inter = fmaxf(ix2 - ix1, 0.0f) * fmaxf(iy2 - iy1, 0.0f);
                float a2 = (ox2 - ox1) * (oy2 - oy1);
                float iou = inter / fmaxf(a1 + a2 - inter, 1e-8f);
                if (iou > 0.5f) ssh[j] = -INFINITY;
            }
            if (t == 0) ssh[bi_sh] = -INFINITY;
        }
        __syncthreads();
    }
}

// ---------------------------------------------------------------------------
// Host launcher
// ---------------------------------------------------------------------------
extern "C" void kernel_launch(void* const* d_in, const int* in_sizes, int n_in,
                              void* d_out, int out_size, void* d_ws, size_t ws_size,
                              hipStream_t stream) {
    const float* features = (const float*)d_in[0];
    const float* proposals = (const float*)d_in[1];
    const float* w1 = (const float*)d_in[2];
    const float* b1 = (const float*)d_in[3];
    const float* w2 = (const float*)d_in[4];
    const float* b2 = (const float*)d_in[5];
    const float* cls_w = (const float*)d_in[6];
    const float* cls_b = (const float*)d_in[7];
    const float* bbox_w = (const float*)d_in[8];
    const float* bbox_b = (const float*)d_in[9];
    float* out = (float*)d_out;

    char* ws = (char*)d_ws;
    size_t off = 0;
    __bf16* Xp = (__bf16*)(ws + off); off += (size_t)NBOX * DIN * 2;   // packed
    __bf16* H1 = (__bf16*)(ws + off); off += (size_t)NBOX * FDIM * 2;  // packed
    __bf16* H2 = (__bf16*)(ws + off); off += (size_t)NBOX * FDIM * 2;  // row-major
    float* logits = (float*)(ws + off); off += (size_t)NBOX * KCLS * 4;
    float* deltas = (float*)(ws + off); off += (size_t)NBOX * KCLS * 4 * 4;
    float* scall  = (float*)(ws + off); off += (size_t)NCAND * 4;
    float* smask  = (float*)(ws + off); off += (size_t)NCAND * 4;
    float* boxesb = (float*)(ws + off); off += (size_t)NCAND * 4 * 4;
    float* oboxb  = (float*)(ws + off); off += (size_t)NCAND * 4 * 4;

    // 1) ROI align -> packed bf16 activations
    roi_align_kernel<<<NBOX, 256, 0, stream>>>(features, proposals, Xp);

    // 2) FC1: full-M blocks, weights streamed from HBM exactly once
    gemm_bf16_wmma<true><<<FDIM / 32, 256, 0, stream>>>(
        Xp, w1, b1, H1, DIN, FDIM);

    // 3) FC2
    gemm_bf16_wmma<false><<<FDIM / 32, 256, 0, stream>>>(
        H1, w2, b2, H2, FDIM, FDIM);

    // 4) heads
    heads_kernel<<<NBOX, 128, 0, stream>>>(H2, cls_w, cls_b, bbox_w, bbox_b,
                                           logits, deltas);

    // 5) softmax/decode/clip
    decode_kernel<<<(NCAND + 255) / 256, 256, 0, stream>>>(
        logits, deltas, proposals, scall, smask, boxesb, oboxb);

    // 6) NMS -> outputs
    nms_kernel<<<1, 1024, 0, stream>>>(scall, smask, boxesb, oboxb, out);
}
